// Memory_Attention_64089501991501
// MI455X (gfx1250) — matrix operand order repaired and checked by reading, not compiled, not run on hardware
//
#include <hip/hip_runtime.h>

#define BB 8
#define LL 4096
#define DD 1024
#define MM 512

typedef __attribute__((ext_vector_type(16))) __bf16 v16bf;
typedef __attribute__((ext_vector_type(8)))  float  v8f;
typedef __attribute__((ext_vector_type(4)))  unsigned v4u;

typedef unsigned short ushort_t;

union BFr { v16bf v; unsigned u[8]; };

// ---- conversions (used only OUTSIDE hot GEMM loops) ----
static __device__ __forceinline__ unsigned short f2bf(float f) {
  unsigned u = __float_as_uint(f);
  u += 0x7fffu + ((u >> 16) & 1u);
  return (unsigned short)(u >> 16);
}
static __device__ __forceinline__ float bf2f(unsigned short h) {
  return __uint_as_float(((unsigned)h) << 16);
}
static __device__ __forceinline__ unsigned pack_bf16pair(float a, float b) {
#if __has_builtin(__builtin_amdgcn_cvt_pk_bf16_f32)
  typedef __attribute__((ext_vector_type(2))) __bf16 v2bf;
  union { v2bf v; unsigned u; } cv;
  cv.v = __builtin_amdgcn_cvt_pk_bf16_f32(a, b);
  return cv.u;
#else
  return (unsigned)f2bf(a) | ((unsigned)f2bf(b) << 16);
#endif
}

// ---------------- Prep: split qs into bf16 hi/lo, row-major AND transposed ----------------
// grid (D/32, L/32, B), block 256. 32x32 fp32 tile; transpose via padded LDS.
__global__ __launch_bounds__(256) void k_prep_q(const float* __restrict__ qs,
                                                ushort_t* __restrict__ qhi,  ushort_t* __restrict__ qlo,
                                                ushort_t* __restrict__ qthi, ushort_t* __restrict__ qtlo) {
  __shared__ ushort_t shi[32][33];
  __shared__ ushort_t slo[32][33];
  const int tx = threadIdx.x & 31;        // d within tile
  const int ty = threadIdx.x >> 5;        // 0..7
  const int b  = blockIdx.z;
  const int l0 = blockIdx.y * 32;
  const int d0 = blockIdx.x * 32;

  #pragma unroll
  for (int j = 0; j < 4; ++j) {
    const int l = ty + j * 8;
    const size_t idx = ((size_t)b * LL + l0 + l) * DD + d0 + tx;
    const float f = qs[idx];
    const unsigned short h = f2bf(f);
    const unsigned short g = f2bf(f - bf2f(h));
    qhi[idx] = h; qlo[idx] = g;
    shi[tx][l] = h; slo[tx][l] = g;      // [d_local][l_local]
  }
  __syncthreads();
  #pragma unroll
  for (int j = 0; j < 4; ++j) {
    const int d = ty + j * 8;
    const size_t idx = ((size_t)b * DD + d0 + d) * LL + l0 + tx;
    qthi[idx] = shi[d][tx];
    qtlo[idx] = slo[d][tx];
  }
}

// ---------------- Prep: split memory into bf16 hi/lo ----------------
__global__ __launch_bounds__(256) void k_prep_mem(const float* __restrict__ mem,
                                                  ushort_t* __restrict__ mhi, ushort_t* __restrict__ mlo) {
  const int i = blockIdx.x * 256 + threadIdx.x;
  const float f = mem[i];
  const unsigned short h = f2bf(f);
  mhi[i] = h;
  mlo[i] = f2bf(f - bf2f(h));
}

// ---------------- Pass 1: S[b] = Mem(512x1024) . Q[b](4096x1024)^T ----------------
// grid (L/256, M/64, B), block 256 (8 waves). Wave strip: 32(m) x 64(l).
__global__ __launch_bounds__(256) void k_scores(const ushort_t* __restrict__ qhi, const ushort_t* __restrict__ qlo,
                                                const ushort_t* __restrict__ mhi, const ushort_t* __restrict__ mlo,
                                                float* __restrict__ S) {
  const int lane = threadIdx.x & 31;
  const int wave = threadIdx.x >> 5;
  const int msub = wave >> 2;
  const int lsub = wave & 3;
  const int b    = blockIdx.z;
  const int mb   = blockIdx.y * 64 + msub * 32;
  const int l0   = blockIdx.x * 256 + lsub * 64;
  const int lr   = lane & 15;
  const int hs   = lane >> 4;

  v8f acc[2][4] = {};
  const ushort_t* ar0h = mhi + (size_t)(mb + lr) * DD;
  const ushort_t* ar0l = mlo + (size_t)(mb + lr) * DD;
  const ushort_t* ar1h = mhi + (size_t)(mb + 16 + lr) * DD;
  const ushort_t* ar1l = mlo + (size_t)(mb + 16 + lr) * DD;
  const size_t qbase = (size_t)b * LL * DD;

  for (int k0 = 0; k0 < DD; k0 += 32) {
    BFr Ah[2], Al[2];
    *(v4u*)&Ah[0].u[0] = *(const v4u*)(ar0h + k0 + hs * 8);
    *(v4u*)&Ah[0].u[4] = *(const v4u*)(ar0h + k0 + 16 + hs * 8);
    *(v4u*)&Al[0].u[0] = *(const v4u*)(ar0l + k0 + hs * 8);
    *(v4u*)&Al[0].u[4] = *(const v4u*)(ar0l + k0 + 16 + hs * 8);
    *(v4u*)&Ah[1].u[0] = *(const v4u*)(ar1h + k0 + hs * 8);
    *(v4u*)&Ah[1].u[4] = *(const v4u*)(ar1h + k0 + 16 + hs * 8);
    *(v4u*)&Al[1].u[0] = *(const v4u*)(ar1l + k0 + hs * 8);
    *(v4u*)&Al[1].u[4] = *(const v4u*)(ar1l + k0 + 16 + hs * 8);

    #pragma unroll
    for (int t = 0; t < 4; ++t) {
      const size_t roff = qbase + (size_t)(l0 + t * 16 + lr) * DD + k0 + hs * 8;
      BFr Bh, Bl;
      *(v4u*)&Bh.u[0] = *(const v4u*)(qhi + roff);
      *(v4u*)&Bh.u[4] = *(const v4u*)(qhi + roff + 16);
      *(v4u*)&Bl.u[0] = *(const v4u*)(qlo + roff);
      *(v4u*)&Bl.u[4] = *(const v4u*)(qlo + roff + 16);
      #pragma unroll
      for (int mi = 0; mi < 2; ++mi) {
        acc[mi][t] = __builtin_amdgcn_wmma_f32_16x16x32_bf16(false, Ah[mi].v, false, Bh.v, (short)0, acc[mi][t], false, false);
        acc[mi][t] = __builtin_amdgcn_wmma_f32_16x16x32_bf16(false, Ah[mi].v, false, Bl.v, (short)0, acc[mi][t], false, false);
        acc[mi][t] = __builtin_amdgcn_wmma_f32_16x16x32_bf16(false, Al[mi].v, false, Bh.v, (short)0, acc[mi][t], false, false);
      }
    }
  }
  #pragma unroll
  for (int mi = 0; mi < 2; ++mi)
    #pragma unroll
    for (int t = 0; t < 4; ++t) {
      const int col = l0 + t * 16 + lr;
      #pragma unroll
      for (int v = 0; v < 8; ++v) {
        const int row = mb + mi * 16 + v + hs * 8;
        S[((size_t)b * MM + row) * LL + col] = acc[mi][t][v];
      }
    }
}

// ---------------- Pass 2: softmax over L; writes bf16 hi/lo probs ----------------
__global__ __launch_bounds__(256) void k_softmax(const float* __restrict__ S,
                                                 ushort_t* __restrict__ phi, ushort_t* __restrict__ plo) {
  __shared__ float red[256];
  const int t = threadIdx.x;
  const size_t rb = (size_t)blockIdx.x * LL;
  float v[16];
  float mx = -__builtin_inff();
  #pragma unroll
  for (int i = 0; i < 16; ++i) { v[i] = S[rb + t + i * 256]; mx = fmaxf(mx, v[i]); }
  red[t] = mx; __syncthreads();
  for (int s = 128; s > 0; s >>= 1) { if (t < s) red[t] = fmaxf(red[t], red[t + s]); __syncthreads(); }
  mx = red[0]; __syncthreads();
  float sum = 0.f;
  #pragma unroll
  for (int i = 0; i < 16; ++i) { v[i] = __expf(v[i] - mx); sum += v[i]; }
  red[t] = sum; __syncthreads();
  for (int s = 128; s > 0; s >>= 1) { if (t < s) red[t] += red[t + s]; __syncthreads(); }
  const float inv = 1.0f / red[0];
  #pragma unroll
  for (int i = 0; i < 16; ++i) {
    const float p = v[i] * inv;
    const unsigned short h = f2bf(p);
    phi[rb + t + i * 256] = h;
    plo[rb + t + i * 256] = f2bf(p - bf2f(h));
  }
  (void)pack_bf16pair; // keep helper referenced-safe
}

// ---------------- Pass 3: R[b] = P[b](512x4096) . Q[b](4096x1024) ----------------
// grid (D/256, M/64, B), block 256. Wave strip: 32(m) x 64(d). B-frags from qt (K=l contiguous).
__global__ __launch_bounds__(256) void k_out(const ushort_t* __restrict__ phi, const ushort_t* __restrict__ plo,
                                             const ushort_t* __restrict__ qthi, const ushort_t* __restrict__ qtlo,
                                             float* __restrict__ R) {
  const int lane = threadIdx.x & 31;
  const int wave = threadIdx.x >> 5;
  const int msub = wave >> 2;
  const int dsub = wave & 3;
  const int b    = blockIdx.z;
  const int mb   = blockIdx.y * 64 + msub * 32;
  const int d0   = blockIdx.x * 256 + dsub * 64;
  const int lr   = lane & 15;
  const int hs   = lane >> 4;

  v8f acc[2][4] = {};
  const ushort_t* pr0h = phi + ((size_t)b * MM + mb + lr) * LL;
  const ushort_t* pr0l = plo + ((size_t)b * MM + mb + lr) * LL;
  const ushort_t* pr1h = phi + ((size_t)b * MM + mb + 16 + lr) * LL;
  const ushort_t* pr1l = plo + ((size_t)b * MM + mb + 16 + lr) * LL;
  const size_t qtbase = (size_t)b * DD * LL;

  for (int l0 = 0; l0 < LL; l0 += 32) {
    BFr Ah[2], Al[2];
    *(v4u*)&Ah[0].u[0] = *(const v4u*)(pr0h + l0 + hs * 8);
    *(v4u*)&Ah[0].u[4] = *(const v4u*)(pr0h + l0 + 16 + hs * 8);
    *(v4u*)&Al[0].u[0] = *(const v4u*)(pr0l + l0 + hs * 8);
    *(v4u*)&Al[0].u[4] = *(const v4u*)(pr0l + l0 + 16 + hs * 8);
    *(v4u*)&Ah[1].u[0] = *(const v4u*)(pr1h + l0 + hs * 8);
    *(v4u*)&Ah[1].u[4] = *(const v4u*)(pr1h + l0 + 16 + hs * 8);
    *(v4u*)&Al[1].u[0] = *(const v4u*)(pr1l + l0 + hs * 8);
    *(v4u*)&Al[1].u[4] = *(const v4u*)(pr1l + l0 + 16 + hs * 8);

    #pragma unroll
    for (int t = 0; t < 4; ++t) {
      const size_t roff = qtbase + (size_t)(d0 + t * 16 + lr) * LL + l0 + hs * 8;
      BFr Bh, Bl;
      *(v4u*)&Bh.u[0] = *(const v4u*)(qthi + roff);
      *(v4u*)&Bh.u[4] = *(const v4u*)(qthi + roff + 16);
      *(v4u*)&Bl.u[0] = *(const v4u*)(qtlo + roff);
      *(v4u*)&Bl.u[4] = *(const v4u*)(qtlo + roff + 16);
      #pragma unroll
      for (int mi = 0; mi < 2; ++mi) {
        acc[mi][t] = __builtin_amdgcn_wmma_f32_16x16x32_bf16(false, Ah[mi].v, false, Bh.v, (short)0, acc[mi][t], false, false);
        acc[mi][t] = __builtin_amdgcn_wmma_f32_16x16x32_bf16(false, Ah[mi].v, false, Bl.v, (short)0, acc[mi][t], false, false);
        acc[mi][t] = __builtin_amdgcn_wmma_f32_16x16x32_bf16(false, Al[mi].v, false, Bh.v, (short)0, acc[mi][t], false, false);
      }
    }
  }
  #pragma unroll
  for (int mi = 0; mi < 2; ++mi)
    #pragma unroll
    for (int t = 0; t < 4; ++t) {
      const int col = d0 + t * 16 + lr;
      #pragma unroll
      for (int v = 0; v < 8; ++v) {
        const int row = mb + mi * 16 + v + hs * 8;
        R[((size_t)b * MM + row) * DD + col] = acc[mi][t][v];
      }
    }
}

extern "C" void kernel_launch(void* const* d_in, const int* in_sizes, int n_in,
                              void* d_out, int out_size, void* d_ws, size_t ws_size,
                              hipStream_t stream) {
  (void)in_sizes; (void)n_in; (void)out_size; (void)ws_size;
  const float* qs  = (const float*)d_in[0];   // [8, 4096, 1024] fp32
  const float* mem = (const float*)d_in[1];   // [512, 1024] fp32
  float* out = (float*)d_out;

  // workspace layout (bytes); total ~386 MB
  char* ws = (char*)d_ws;
  const size_t SML  = (size_t)BB * MM * LL;   // 16M elems
  const size_t SLD  = (size_t)BB * LL * DD;   // 33.5M elems
  float*    S    = (float*)   (ws);                                   // 64 MB
  ushort_t* phi  = (ushort_t*)(ws + SML * 4);                         // 32 MB
  ushort_t* plo  = (ushort_t*)(ws + SML * 4 + SML * 2);               // 32 MB
  ushort_t* qhi  = (ushort_t*)(ws + SML * 8);                         // 64 MB
  ushort_t* qlo  = (ushort_t*)(ws + SML * 8 + SLD * 2);               // 64 MB
  ushort_t* qthi = (ushort_t*)(ws + SML * 8 + SLD * 4);               // 64 MB
  ushort_t* qtlo = (ushort_t*)(ws + SML * 8 + SLD * 6);               // 64 MB
  ushort_t* mhi  = (ushort_t*)(ws + SML * 8 + SLD * 8);               // 1 MB
  ushort_t* mlo  = (ushort_t*)(ws + SML * 8 + SLD * 8 + (size_t)MM * DD * 2);

  dim3 blk(256);
  k_prep_q  <<<dim3(DD / 32, LL / 32, BB), blk, 0, stream>>>(qs, qhi, qlo, qthi, qtlo);
  k_prep_mem<<<dim3((MM * DD) / 256),      blk, 0, stream>>>(mem, mhi, mlo);
  k_scores  <<<dim3(LL / 256, MM / 64, BB), blk, 0, stream>>>(qhi, qlo, mhi, mlo, S);
  k_softmax <<<dim3(BB * MM),               blk, 0, stream>>>(S, phi, plo);
  k_out     <<<dim3(DD / 256, MM / 64, BB), blk, 0, stream>>>(phi, plo, qthi, qtlo, out);

  hipMemcpyAsync(out + (size_t)BB * MM * DD, mem, (size_t)MM * DD * sizeof(float),
                 hipMemcpyDeviceToDevice, stream);
}
